// CausalSelfAttention_61761629716702
// MI455X (gfx1250) — compile-verified, matrix operand order checked by reading, not run on hardware
//
#include <hip/hip_runtime.h>
#include <math.h>

// CausalSelfAttention for MI455X (gfx1250, wave32, WMMA).
// All matmuls (QKV proj, QK^T, PV, out proj) run on v_wmma_f32_16x16x32_f16.
// Roofline: ~220 GFLOP vs ~100MB traffic -> matrix-core bound, so everything
// is staged to f16 and pushed through the tensor pipes. Flash attention stages
// K/V tiles through LDS with double-buffered global_load_async_to_lds_b128
// (ASYNCcnt) so copies overlap WMMA+softmax and the 4 waves share each tile.

#define DIMX  2048
#define SEQ   4096
#define NH    16
#define NKV   4
#define HD    128
#define KVDIM (NKV * HD)   // 512

typedef __attribute__((ext_vector_type(16))) _Float16 v16h;
typedef __attribute__((ext_vector_type(8)))  _Float16 h8;
typedef __attribute__((ext_vector_type(8)))  float    v8f;

// ---------------------------------------------------------------------------
// Fragment loader for a 16x32 f16 tile out of a row-major matrix (ld in elems).
// Per CDNA5 ISA 16-bit A layout: lane l holds row (l&15); k groups
// [(l>>4)*8 .. +8) and [16+(l>>4)*8 .. +8)  -> two contiguous 16B loads.
// Works for both A (rows of X) and B (rows of W, since C = X * W^T).
// ---------------------------------------------------------------------------
__device__ __forceinline__ v16h load_frag(const _Float16* __restrict__ tile, int ld) {
  int lane = threadIdx.x & 31;
  int row  = lane & 15;
  int cb   = (lane >> 4) << 3;
  const _Float16* p = tile + row * ld + cb;
  h8 lo = *(const h8*)(p);
  h8 hi = *(const h8*)(p + 16);
  v16h a;
#pragma unroll
  for (int i = 0; i < 8; ++i) { a[i] = lo[i]; a[8 + i] = hi[i]; }
  return a;
}

__device__ __forceinline__ v16h frag_from(const h8 lo, const h8 hi) {
  v16h a;
#pragma unroll
  for (int i = 0; i < 8; ++i) { a[i] = lo[i]; a[8 + i] = hi[i]; }
  return a;
}

// Async copy of 16B/lane from global to LDS (GLOBAL_LOAD_ASYNC_TO_LDS_B128,
// tracked by ASYNCcnt). lds_off = LDS byte offset (low 32 bits of generic ptr).
__device__ __forceinline__ void async_copy_b128(unsigned lds_off, const void* gaddr) {
  asm volatile("global_load_async_to_lds_b128 %0, %1, off"
               :: "v"(lds_off), "v"((unsigned long long)(uintptr_t)gaddr)
               : "memory");
}

// ---------------------------------------------------------------------------
// BitLinear weight quantization: per-row s = clip(mean|W|,1e-5),
// w = clip(round(W/s),-1,1)*s, stored as f16.
// ---------------------------------------------------------------------------
__global__ void __launch_bounds__(256)
quantize_w(const float* __restrict__ W, _Float16* __restrict__ out, int cols) {
  __shared__ float red[256];
  int r = blockIdx.x, tid = threadIdx.x;
  float s = 0.f;
  for (int c = tid; c < cols; c += 256) s += fabsf(W[r * cols + c]);
  red[tid] = s;
  __syncthreads();
  for (int st = 128; st > 0; st >>= 1) {
    if (tid < st) red[tid] += red[tid + st];
    __syncthreads();
  }
  float sc = fmaxf(red[0] / (float)cols, 1e-5f);
  for (int c = tid; c < cols; c += 256) {
    float q = fminf(fmaxf(rintf(W[r * cols + c] / sc), -1.f), 1.f);
    out[r * cols + c] = (_Float16)(q * sc);
  }
}

__global__ void __launch_bounds__(256)
cast_f32_f16(const float* __restrict__ in, _Float16* __restrict__ out) {
  int i = blockIdx.x * 256 + threadIdx.x;
  out[i] = (_Float16)in[i];
}

// ---------------------------------------------------------------------------
// GEMM: C[M][N] (f32) = A[M][K] (f16, row-major) * B[N][K]^T (f16, row-major).
// 8 waves / block, each wave computes a 64x64 tile (4x4 WMMA accumulators).
// ---------------------------------------------------------------------------
__global__ void __launch_bounds__(256)
gemm_xwt(const _Float16* __restrict__ A, const _Float16* __restrict__ B,
         float* __restrict__ C, int M, int N, int K) {
  int wave   = threadIdx.x >> 5;
  int gid    = blockIdx.x * 8 + wave;
  int tilesM = M >> 6;
  int tm = (gid % tilesM) * 64;
  int tn = (gid / tilesM) * 64;
  int lane = threadIdx.x & 31;

  v8f acc[4][4] = {};
  for (int k0 = 0; k0 < K; k0 += 32) {
    v16h a[4], b[4];
#pragma unroll
    for (int i = 0; i < 4; ++i) a[i] = load_frag(A + (tm + i * 16) * K + k0, K);
#pragma unroll
    for (int i = 0; i < 4; ++i) b[i] = load_frag(B + (tn + i * 16) * K + k0, K);
    if (k0 + 32 < K) {  // global_prefetch_b8 for the next K-slab
      __builtin_prefetch(A + (tm + (lane & 15)) * K + k0 + 32, 0, 1);
      __builtin_prefetch(B + (tn + (lane & 15)) * K + k0 + 32, 0, 1);
    }
#pragma unroll
    for (int mi = 0; mi < 4; ++mi)
#pragma unroll
      for (int ni = 0; ni < 4; ++ni)
        acc[mi][ni] = __builtin_amdgcn_wmma_f32_16x16x32_f16(
            false, a[mi], false, b[ni], (short)0, acc[mi][ni], false, false);
  }
  int m0 = (lane >> 4) * 8, n0 = lane & 15;
#pragma unroll
  for (int mi = 0; mi < 4; ++mi)
#pragma unroll
    for (int ni = 0; ni < 4; ++ni)
#pragma unroll
      for (int j = 0; j < 8; ++j)
        C[(tm + mi * 16 + m0 + j) * N + (tn + ni * 16 + n0)] = acc[mi][ni][j];
}

// ---------------------------------------------------------------------------
// Q post: RMS-norm over HD, RoPE, * q_gain[h] * 1/sqrt(HD); -> qb[h][t][d] f16
// ---------------------------------------------------------------------------
__global__ void __launch_bounds__(64)
post_q(const float* __restrict__ qf, const float* __restrict__ gain,
       _Float16* __restrict__ qb) {
  __shared__ float red[64];
  int h = blockIdx.x, t = blockIdx.y, d = threadIdx.x;
  float x1 = qf[t * DIMX + h * HD + d];
  float x2 = qf[t * DIMX + h * HD + d + 64];
  red[d] = x1 * x1 + x2 * x2;
  __syncthreads();
  for (int st = 32; st > 0; st >>= 1) {
    if (d < st) red[d] += red[d + st];
    __syncthreads();
  }
  float r  = rsqrtf(red[0] / (float)HD + 1.1920929e-07f);
  float fr = (float)t * __expf(-(float)d * (logf(10000.0f) / 64.0f));
  float c = cosf(fr), sn = sinf(fr);
  float g = gain[h] * 0.08838834764831845f;  // fold in 1/sqrt(HD)
  qb[(h * SEQ + t) * HD + d]      = (_Float16)((x1 * c + x2 * sn) * r * g);
  qb[(h * SEQ + t) * HD + d + 64] = (_Float16)((-x1 * sn + x2 * c) * r * g);
}

// K post: RMS-norm + RoPE -> kb[kvh][t][d] f16
__global__ void __launch_bounds__(64)
post_k(const float* __restrict__ kf, _Float16* __restrict__ kb) {
  __shared__ float red[64];
  int h = blockIdx.x, t = blockIdx.y, d = threadIdx.x;
  float x1 = kf[t * KVDIM + h * HD + d];
  float x2 = kf[t * KVDIM + h * HD + d + 64];
  red[d] = x1 * x1 + x2 * x2;
  __syncthreads();
  for (int st = 32; st > 0; st >>= 1) {
    if (d < st) red[d] += red[d + st];
    __syncthreads();
  }
  float r  = rsqrtf(red[0] / (float)HD + 1.1920929e-07f);
  float fr = (float)t * __expf(-(float)d * (logf(10000.0f) / 64.0f));
  float c = cosf(fr), sn = sinf(fr);
  kb[(h * SEQ + t) * HD + d]      = (_Float16)((x1 * c + x2 * sn) * r);
  kb[(h * SEQ + t) * HD + d + 64] = (_Float16)((-x1 * sn + x2 * c) * r);
}

// V post: transpose to vT[kvh][d][t] f16 so P*V uses the same B-frag loader.
__global__ void __launch_bounds__(256)
post_v(const float* __restrict__ vf, _Float16* __restrict__ vT) {
  int idx = blockIdx.x * 256 + threadIdx.x;  // over SEQ*KVDIM
  int t = idx / KVDIM, c = idx % KVDIM;
  int kvh = c / HD, d = c % HD;
  vT[kvh * (HD * SEQ) + d * SEQ + t] = (_Float16)vf[idx];
}

// ---------------------------------------------------------------------------
// Flash attention (causal, GQA). Block = 4 waves; wave owns a 16-row Q strip.
// KV blocks of 64. K/V tiles are staged to LDS with double-buffered async
// copies (ASYNCcnt) shared by all 4 waves; S = Q K^T via WMMA, streaming
// softmax with shfl_xor row reductions, P staged through LDS (s_wait_dscnt)
// and reloaded as A frags, O += P V via WMMA against transposed V.
// ---------------------------------------------------------------------------
__global__ void __launch_bounds__(128)
flash_attn(const _Float16* __restrict__ qb, const _Float16* __restrict__ kb,
           const _Float16* __restrict__ vT, _Float16* __restrict__ yb) {
  __shared__ _Float16 Kt[2][64][HD];   // K tile (seq x d), double buffered: 32KB
  __shared__ _Float16 Vt[2][HD][64];   // V^T tile (d x seq), double buffered: 32KB
  __shared__ _Float16 Pl[4][16][64];   // per-wave P staging: 8KB

  int tid  = threadIdx.x;
  int wave = tid >> 5, lane = tid & 31;
  int h = blockIdx.y, kvh = h >> 2;
  int qt = blockIdx.x;
  int q0 = qt * 64 + wave * 16;
  int n0 = lane & 15, mh = lane >> 4;
  int row = lane & 15, cb = mh * 8;

  const _Float16* Kbase = kb + (size_t)kvh * SEQ * HD;
  const _Float16* Vbase = vT + (size_t)kvh * HD * SEQ;

  // Q fragments live in registers for the whole block sweep.
  const _Float16* Q = qb + (h * SEQ + q0) * HD;
  v16h qa[4];
#pragma unroll
  for (int f = 0; f < 4; ++f) qa[f] = load_frag(Q + f * 32, HD);

  // Cooperative async staging of one KV block: K tile 64x128 f16 (16KB) and
  // V^T tile 128x64 f16 (16KB); 128 threads x 8 chunks x 16B each.
  auto stage = [&](int jb, int buf) {
    const _Float16* Kp = Kbase + jb * 64 * HD;
    const _Float16* Vp = Vbase + jb * 64;
#pragma unroll
    for (int c = 0; c < 8; ++c) {
      int ck = c * 128 + tid;
      int kr = ck >> 4, kc = (ck & 15) * 8;   // 16 chunks per 128-wide K row
      async_copy_b128((unsigned)(uintptr_t)&Kt[buf][kr][kc], Kp + kr * HD + kc);
      int vr = ck >> 3, vc = (ck & 7) * 8;    // 8 chunks per 64-wide V row
      async_copy_b128((unsigned)(uintptr_t)&Vt[buf][vr][vc], Vp + vr * SEQ + vc);
    }
  };

  stage(0, 0);

  v8f o[8] = {};
  float mrun[8], lrun[8];
#pragma unroll
  for (int j = 0; j < 8; ++j) { mrun[j] = -3.0e38f; lrun[j] = 0.f; }

  for (int jb = 0; jb <= qt; ++jb) {
    int buf = jb & 1;
    asm volatile("s_wait_asynccnt 0" ::: "memory");  // own async copies done
    __syncthreads();                                 // all waves' copies visible
    if (jb < qt) stage(jb + 1, buf ^ 1);             // overlap next tile copy

    // S = Q K^T from LDS K tile
    v8f s[4] = {};
#pragma unroll
    for (int nt = 0; nt < 4; ++nt)
#pragma unroll
      for (int f = 0; f < 4; ++f) {
        h8 lo = *(const h8*)&Kt[buf][nt * 16 + row][f * 32 + cb];
        h8 hi = *(const h8*)&Kt[buf][nt * 16 + row][f * 32 + cb + 16];
        s[nt] = __builtin_amdgcn_wmma_f32_16x16x32_f16(
            false, qa[f], false, frag_from(lo, hi), (short)0, s[nt], false, false);
      }
    if (jb == qt) {  // causal mask on the diagonal block
#pragma unroll
      for (int nt = 0; nt < 4; ++nt)
#pragma unroll
        for (int j = 0; j < 8; ++j)
          if (jb * 64 + nt * 16 + n0 > q0 + mh * 8 + j) s[nt][j] = -3.0e38f;
    }
    float nm[8], al[8];
#pragma unroll
    for (int j = 0; j < 8; ++j) {
      float mx = fmaxf(fmaxf(s[0][j], s[1][j]), fmaxf(s[2][j], s[3][j]));
      mx = fmaxf(mx, __shfl_xor(mx, 1, 32));
      mx = fmaxf(mx, __shfl_xor(mx, 2, 32));
      mx = fmaxf(mx, __shfl_xor(mx, 4, 32));
      mx = fmaxf(mx, __shfl_xor(mx, 8, 32));
      nm[j] = fmaxf(mrun[j], mx);
      al[j] = __expf(mrun[j] - nm[j]);
      mrun[j] = nm[j];
      lrun[j] *= al[j];
    }
#pragma unroll
    for (int d = 0; d < 8; ++d)
#pragma unroll
      for (int j = 0; j < 8; ++j) o[d][j] *= al[j];
#pragma unroll
    for (int j = 0; j < 8; ++j) {
      float ps = 0.f;
#pragma unroll
      for (int nt = 0; nt < 4; ++nt) {
        float p = __expf(s[nt][j] - nm[j]);
        Pl[wave][mh * 8 + j][nt * 16 + n0] = (_Float16)p;
        ps += p;
      }
      ps += __shfl_xor(ps, 1, 32);
      ps += __shfl_xor(ps, 2, 32);
      ps += __shfl_xor(ps, 4, 32);
      ps += __shfl_xor(ps, 8, 32);
      lrun[j] += ps;
    }
    asm volatile("s_wait_dscnt 0" ::: "memory");  // P stores visible before reload
    v16h pa[2];
#pragma unroll
    for (int f = 0; f < 2; ++f) {
      h8 lo = *(const h8*)&Pl[wave][row][f * 32 + cb];
      h8 hi = *(const h8*)&Pl[wave][row][f * 32 + cb + 16];
      pa[f] = frag_from(lo, hi);
    }
    // O += P V from LDS V^T tile
#pragma unroll
    for (int d = 0; d < 8; ++d)
#pragma unroll
      for (int f = 0; f < 2; ++f) {
        h8 lo = *(const h8*)&Vt[buf][d * 16 + row][f * 32 + cb];
        h8 hi = *(const h8*)&Vt[buf][d * 16 + row][f * 32 + cb + 16];
        o[d] = __builtin_amdgcn_wmma_f32_16x16x32_f16(
            false, pa[f], false, frag_from(lo, hi), (short)0, o[d], false, false);
      }
  }
#pragma unroll
  for (int j = 0; j < 8; ++j) {
    float inv = 1.0f / lrun[j];
    int tok = q0 + mh * 8 + j;
#pragma unroll
    for (int d = 0; d < 8; ++d)
      yb[tok * DIMX + h * HD + d * 16 + n0] = (_Float16)(o[d][j] * inv);
  }
}

// ---------------------------------------------------------------------------
extern "C" void kernel_launch(void* const* d_in, const int* in_sizes, int n_in,
                              void* d_out, int out_size, void* d_ws, size_t ws_size,
                              hipStream_t stream) {
  const float* x    = (const float*)d_in[0];
  const float* wq   = (const float*)d_in[1];
  const float* wk   = (const float*)d_in[2];
  const float* wv   = (const float*)d_in[3];
  const float* wp   = (const float*)d_in[4];
  const float* gain = (const float*)d_in[5];
  float* out = (float*)d_out;

  char* w = (char*)d_ws;
  _Float16* xb  = (_Float16*)w;                 w += (size_t)SEQ * DIMX * 2;
  _Float16* wqq = (_Float16*)w;                 w += (size_t)DIMX * DIMX * 2;
  _Float16* wkq = (_Float16*)w;                 w += (size_t)KVDIM * DIMX * 2;
  _Float16* wvq = (_Float16*)w;                 w += (size_t)KVDIM * DIMX * 2;
  _Float16* wpq = (_Float16*)w;                 w += (size_t)DIMX * DIMX * 2;
  float*    qf  = (float*)w;                    w += (size_t)SEQ * DIMX * 4;
  float*    kf  = (float*)w;                    w += (size_t)SEQ * KVDIM * 4;
  float*    vf  = (float*)w;                    w += (size_t)SEQ * KVDIM * 4;
  _Float16* qb  = (_Float16*)w;                 w += (size_t)SEQ * DIMX * 2;
  _Float16* kbf = (_Float16*)w;                 w += (size_t)NKV * SEQ * HD * 2;
  _Float16* vT  = (_Float16*)w;                 w += (size_t)NKV * HD * SEQ * 2;
  _Float16* yb  = (_Float16*)w;                 w += (size_t)SEQ * DIMX * 2;
  (void)ws_size; (void)in_sizes; (void)n_in; (void)out_size;

  // 1) BitLinear weight quantization + x cast to f16
  quantize_w<<<DIMX,  256, 0, stream>>>(wq, wqq, DIMX);
  quantize_w<<<KVDIM, 256, 0, stream>>>(wk, wkq, DIMX);
  quantize_w<<<KVDIM, 256, 0, stream>>>(wv, wvq, DIMX);
  quantize_w<<<DIMX,  256, 0, stream>>>(wp, wpq, DIMX);
  cast_f32_f16<<<(SEQ * DIMX) / 256, 256, 0, stream>>>(x, xb);

  // 2) Projections via WMMA GEMM (C = X W^T)
  gemm_xwt<<<(SEQ / 64) * (DIMX / 64) / 8,  256, 0, stream>>>(xb, wqq, qf, SEQ, DIMX,  DIMX);
  gemm_xwt<<<(SEQ / 64) * (KVDIM / 64) / 8, 256, 0, stream>>>(xb, wkq, kf, SEQ, KVDIM, DIMX);
  gemm_xwt<<<(SEQ / 64) * (KVDIM / 64) / 8, 256, 0, stream>>>(xb, wvq, vf, SEQ, KVDIM, DIMX);

  // 3) RMS-norm + RoPE (+gain*scale) + layout conversion to f16
  post_q<<<dim3(NH,  SEQ), 64, 0, stream>>>(qf, gain, qb);
  post_k<<<dim3(NKV, SEQ), 64, 0, stream>>>(kf, kbf);
  post_v<<<(SEQ * KVDIM) / 256, 256, 0, stream>>>(vf, vT);

  // 4) Causal flash attention (WMMA QK^T and PV, async LDS staging)
  flash_attn<<<dim3(SEQ / 64, NH), 128, 0, stream>>>(qb, kbf, vT, yb);

  // 5) Output projection -> f32 result
  gemm_xwt<<<(SEQ / 64) * (DIMX / 64) / 8, 256, 0, stream>>>(yb, wpq, out, SEQ, DIMX, DIMX);
}
